// Neuralsymbol_77154792505932
// MI455X (gfx1250) — matrix operand order repaired and checked by reading, not compiled, not run on hardware
//
#include <hip/hip_runtime.h>
#include <hip/hip_bf16.h>
#include <stdint.h>

// ---------------- problem constants ----------------
#define B_  256
#define T_  128
#define V_  256
#define H_  512
#define NS_ 4
#define MB_ 16          // batch rows per workgroup (= WMMA M)
#define NT_ 512         // threads per workgroup (16 waves)
#define ZO_ 1e-6f
#define EPS_ 1e-12f

typedef __attribute__((ext_vector_type(16))) __bf16 bf16x16;
typedef __attribute__((ext_vector_type(8)))  float  floatx8;

union Frag { uint4 q[2]; bf16x16 v; };

__device__ __forceinline__ uint16_t f2bf(float f) {
  uint32_t u = __float_as_uint(f);
  uint32_t r = u + 0x7FFFu + ((u >> 16) & 1u);   // round-to-nearest-even
  return (uint16_t)(r >> 16);
}

// ---------------- prep kernels (256 threads) ----------------
__global__ void rownorm_bf16_k(const float* __restrict__ src, uint16_t* __restrict__ dst) {
  __shared__ float red[256];
  const int row = blockIdx.x;
  const float v = src[(size_t)row * 256 + threadIdx.x];
  red[threadIdx.x] = v * v;
  __syncthreads();
  for (int st = 128; st > 0; st >>= 1) {
    if (threadIdx.x < st) red[threadIdx.x] += red[threadIdx.x + st];
    __syncthreads();
  }
  const float scale = 1.f / fmaxf(sqrtf(red[0]), EPS_);
  dst[(size_t)row * 256 + threadIdx.x] = f2bf(v * scale);
}

__global__ void rownorm_f32_k(const float* __restrict__ src, float* __restrict__ dst) {
  __shared__ float red[256];
  const int row = blockIdx.x;
  const float v = src[(size_t)row * 256 + threadIdx.x];
  red[threadIdx.x] = v * v;
  __syncthreads();
  for (int st = 128; st > 0; st >>= 1) {
    if (threadIdx.x < st) red[threadIdx.x] += red[threadIdx.x + st];
    __syncthreads();
  }
  const float scale = 1.f / fmaxf(sqrtf(red[0]), EPS_);
  dst[(size_t)row * 256 + threadIdx.x] = v * scale;
}

__global__ void cvt_bf16_k(const float* __restrict__ src, uint16_t* __restrict__ dst, int n) {
  for (int i = blockIdx.x * blockDim.x + threadIdx.x; i < n; i += gridDim.x * blockDim.x)
    dst[i] = f2bf(src[i]);
}

// initial state after stack_push(ZO stack, e0 ptr, init_vec): slot1 = init_vec, ptr = e1
__global__ void stack_init_k(const float* __restrict__ initvec,
                             float* __restrict__ sc, float* __restrict__ sw,
                             float* __restrict__ pc, float* __restrict__ pw) {
  const int idx = blockIdx.x * 256 + threadIdx.x;
  if (idx < B_ * NS_ * V_) {
    const int n = (idx >> 8) & 3;
    const int v = idx & 255;
    const float val = (n == 1) ? initvec[v] : ZO_;
    sc[idx] = val;
    sw[idx] = val;
  }
  if (idx < B_ * NS_) {
    const float pv = ((idx & 3) == 1) ? 1.f : 0.f;
    pc[idx] = pv;
    pw[idx] = pv;
  }
}

// ---------------- device helpers (main kernel, 512 threads) ----------------

// circular conv: out[r][v] = sum_u a[r][u] * b[r][(v-u) mod 256]
// 4-tap blocked; b duplicated into dup[r][0..511] (dup[j]=b[j&255]) so every
// window load is an aligned ds_load_b128.  Each thread: 8 outputs, 2048 FMA,
// 256 b128 LDS loads.
__device__ __forceinline__ void circconv256(const float* __restrict__ a,
                                            const float* __restrict__ b,
                                            float* __restrict__ out, int ldo,
                                            float* __restrict__ dup) {
  const int tid = threadIdx.x;
  const int r   = tid >> 5;
  const int c   = tid & 31;
  // fill dup (each 32-thread group fills its row)
  {
    float* dr = dup + r * 512;
    const float* br = b + r * V_;
#pragma unroll
    for (int k = 0; k < 16; ++k) {
      const int j = c * 16 + k;
      dr[j] = br[j & 255];
    }
  }
  __syncthreads();
  const int v0 = c << 3;                 // 8 outputs per thread
  const float* ar = a + r * V_;
  const float* dr = dup + r * 512;
  float acc[8];
#pragma unroll
  for (int i = 0; i < 8; ++i) acc[i] = 0.f;
  for (int u = 0; u < V_; u += 4) {
    const float4 a4 = *(const float4*)(ar + u);
    const float* w = dr + (v0 - u - 4 + 256);   // 16B-aligned
    float bw[12];
    *(float4*)&bw[0] = *(const float4*)(w);
    *(float4*)&bw[4] = *(const float4*)(w + 4);
    *(float4*)&bw[8] = *(const float4*)(w + 8);
    // tap d: b index = dup[w0 + 4 + i - d]
#pragma unroll
    for (int i = 0; i < 8; ++i)
      acc[i] += a4.x * bw[i + 4] + a4.y * bw[i + 3] + a4.z * bw[i + 2] + a4.w * bw[i + 1];
  }
  float* orow = out + r * ldo + v0;
#pragma unroll
  for (int i = 0; i < 8; ++i) orow[i] = acc[i];
}

__device__ __forceinline__ void l2norm_ip(float* __restrict__ m, float (*red)[72]) {
  const int r = threadIdx.x >> 5, c = threadIdx.x & 31, c0 = c << 3;
  float* row = m + r * V_;
  float s = 0.f;
#pragma unroll
  for (int i = 0; i < 8; ++i) { const float v = row[c0 + i]; s += v * v; }
  red[r][c] = s;
  __syncthreads();
  if (c == 0) {
    float t = 0.f;
    for (int k = 0; k < 32; ++k) t += red[r][k];
    red[r][64] = 1.f / fmaxf(sqrtf(t), EPS_);
  }
  __syncthreads();
  const float sc = red[r][64];
#pragma unroll
  for (int i = 0; i < 8; ++i) row[c0 + i] *= sc;
  __syncthreads();
}

// C(16xN,f32 LDS) = A(16xK bf16 LDS, lda=512) x B^T(K x N from [N][ldb] bf16 global)
// bf16 WMMA 16x16x32, f32 accumulate.  mode 1 => relu on store.  16 waves.
__device__ __forceinline__ void gemm16(const uint16_t* __restrict__ A,
                                       const uint16_t* __restrict__ B, int ldb,
                                       float* __restrict__ C, int ldc,
                                       int N, int K, int mode) {
  const int tid  = threadIdx.x;
  const int wave = tid >> 5;
  const int lane = tid & 31;
  const int l15  = lane & 15;
  const int hi   = lane >> 4;
  const uint16_t* arow = A + l15 * H_ + hi * 8;  // A: lane<16 K{0..7,16..23}, lane>=16 K{8..15,24..31}
  for (int n0 = wave * 16; n0 < N; n0 += 256) {
    floatx8 acc = {0.f, 0.f, 0.f, 0.f, 0.f, 0.f, 0.f, 0.f};
    const uint16_t* brow = B + (size_t)(n0 + l15) * ldb + hi * 8;  // B: lane<16 K{0..7,16..23}, lane>=16 K{8..15,24..31}
    for (int k0 = 0; k0 < K; k0 += 32) {
      Frag a, b;
      a.q[0] = *(const uint4*)(arow + k0);
      a.q[1] = *(const uint4*)(arow + k0 + 16);
      b.q[0] = *(const uint4*)(brow + k0);
      b.q[1] = *(const uint4*)(brow + k0 + 16);
      acc = __builtin_amdgcn_wmma_f32_16x16x32_bf16(false, a.v, false, b.v,
                                                    (short)0, acc, false, false);
    }
    float* crow = C + (size_t)(hi * 8) * ldc + n0 + l15;  // D: col=n0+l15, row=vgpr+hi*8
#pragma unroll
    for (int rr = 0; rr < 8; ++rr) {
      float vv = acc[rr];
      if (mode == 1) vv = vv > 0.f ? vv : 0.f;
      crow[rr * ldc] = vv;
    }
  }
}

// ---------------- main persistent kernel ----------------
__global__ __launch_bounds__(NT_, 1) void ns_main(
    const float* __restrict__ x, const float* __restrict__ W_isa2,
    const float* __restrict__ op_b1, const float* __restrict__ op_W2,
    const float* __restrict__ op_b2, const float* __restrict__ W_sel2,
    const float* __restrict__ out_sym, const float* __restrict__ csharp,
    const float* __restrict__ wsharp,
    const uint16_t* __restrict__ sym_bf, const uint16_t* __restrict__ sel_bf,
    const uint16_t* __restrict__ opw1_bf, const uint16_t* __restrict__ wct_bf,
    const uint16_t* __restrict__ wwk_bf, const float* __restrict__ wisa_n,
    float* __restrict__ sc, float* __restrict__ sw_,
    float* __restrict__ pc, float* __restrict__ pw,
    float* __restrict__ out) {
  __shared__ __align__(16) float    s_x[MB_][V_];     // raw x -> l2norm(x)
  __shared__ __align__(16) float    s_ctrl[MB_][V_];  // control -> l2norm
  __shared__ __align__(16) float    s_work[MB_][V_];  // work -> l2norm
  __shared__ __align__(16) float    s_typ[MB_][V_];   // typ -> l2norm
  __shared__ __align__(16) float    s_conv[MB_][V_];
  __shared__ __align__(16) float    s_newc[MB_][V_];
  __shared__ __align__(16) float    s_neww[MB_][V_];
  __shared__ __align__(16) float    s_dup[MB_][512];  // wrap-free conv window
  __shared__ __align__(16) float    s_h[MB_][H_];
  __shared__ __align__(16) float    s_log[MB_][H_];
  __shared__ __align__(16) uint16_t s_abf[MB_][H_];   // bf16 A-operand buffer
  __shared__ float s_red[MB_][72];
  __shared__ float s_g[MB_][8];
  __shared__ float s_sm[MB_][8];
  __shared__ float s_pt[MB_][32];

  const int tid = threadIdx.x;
  const int b0  = blockIdx.x * MB_;
  const int r   = tid >> 5;
  const int c   = tid & 31;
  const int c0v = c << 3;   // 8-wide chunk over V
  const int c0h = c << 4;   // 16-wide chunk over H

  for (int t = 0; t < T_; ++t) {
    // ---- A/B: load x tile, stack reads ----
    {
      const int bb = b0 + r;
      const float q0 = pc[bb * 4 + 0], q1 = pc[bb * 4 + 1], q2 = pc[bb * 4 + 2], q3 = pc[bb * 4 + 3];
      const float w0 = pw[bb * 4 + 0], w1 = pw[bb * 4 + 1], w2 = pw[bb * 4 + 2], w3 = pw[bb * 4 + 3];
#pragma unroll
      for (int i = 0; i < 8; ++i) {
        const int v = c0v + i;
        s_x[r][v] = x[((size_t)bb * T_ + t) * V_ + v];
        s_ctrl[r][v] = q0 * sc[(size_t)(bb * 4 + 0) * V_ + v] + q1 * sc[(size_t)(bb * 4 + 1) * V_ + v] +
                       q2 * sc[(size_t)(bb * 4 + 2) * V_ + v] + q3 * sc[(size_t)(bb * 4 + 3) * V_ + v];
        s_work[r][v] = w0 * sw_[(size_t)(bb * 4 + 0) * V_ + v] + w1 * sw_[(size_t)(bb * 4 + 1) * V_ + v] +
                       w2 * sw_[(size_t)(bb * 4 + 2) * V_ + v] + w3 * sw_[(size_t)(bb * 4 + 3) * V_ + v];
      }
    }
    __syncthreads();

    // ---- C: typ = tanh(softmax(x @ W_isa_n^T) @ W_isa2^T) ----
    if (tid < MB_ * 4) {
      const int rr = tid >> 2, i = tid & 3;
      const float* wr = wisa_n + i * V_;
      float s = 0.f;
      for (int v = 0; v < V_; ++v) s += s_x[rr][v] * wr[v];
      s_sm[rr][i] = s;
    }
    __syncthreads();
    if (tid < MB_) {
      float m = fmaxf(fmaxf(s_sm[tid][0], s_sm[tid][1]), fmaxf(s_sm[tid][2], s_sm[tid][3]));
      float e[4], se = 0.f;
#pragma unroll
      for (int i = 0; i < 4; ++i) { e[i] = __expf(s_sm[tid][i] - m); se += e[i]; }
      const float inv = 1.f / se;
#pragma unroll
      for (int i = 0; i < 4; ++i) s_sm[tid][i] = e[i] * inv;
    }
    __syncthreads();
#pragma unroll
    for (int i = 0; i < 8; ++i) {
      const int v = c0v + i;
      float s = 0.f;
#pragma unroll
      for (int j = 0; j < 4; ++j) s += s_sm[r][j] * W_isa2[v * 4 + j];
      s_typ[r][v] = tanhf(s);
    }
    __syncthreads();

    // ---- D: l2 normalize x, control, work, typ ----
    l2norm_ip(&s_x[0][0], s_red);
    l2norm_ip(&s_ctrl[0][0], s_red);
    l2norm_ip(&s_work[0][0], s_red);
    l2norm_ip(&s_typ[0][0], s_red);

    // ---- E/F: h0 = circ(ctrl_n, typ_n) @ sym_wn^T ----
    circconv256(&s_ctrl[0][0], &s_typ[0][0], &s_conv[0][0], V_, &s_dup[0][0]);
    __syncthreads();
#pragma unroll
    for (int i = 0; i < 8; ++i) s_abf[r][c0v + i] = f2bf(s_conv[r][c0v + i]);
    __syncthreads();
    gemm16(&s_abf[0][0], sym_bf, V_, &s_h[0][0], H_, H_, V_, 0);
    __syncthreads();
#pragma unroll
    for (int i = 0; i < 16; ++i) s_abf[r][c0h + i] = f2bf(s_h[r][c0h + i]);  // h0 bf16
    __syncthreads();

    // ---- G: six op heads, online softmax over g, gates ----
    for (int o = 0; o < 6; ++o) {
      gemm16(&s_abf[0][0], opw1_bf + (size_t)o * H_ * H_, H_, &s_log[0][0], H_, H_, H_, 0);
      __syncthreads();
      const float* b1r = op_b1 + o * H_;
      const float* w2r = op_W2 + o * H_;
      float lm = -1e30f;
#pragma unroll
      for (int j = 0; j < 16; ++j) lm = fmaxf(lm, s_log[r][c0h + j] + b1r[c0h + j]);
      s_red[r][c] = lm;
      __syncthreads();
      if (c == 0) {
        float m = -1e30f;
        for (int k = 0; k < 32; ++k) m = fmaxf(m, s_red[r][k]);
        s_red[r][64] = m;
      }
      __syncthreads();
      const float m = s_red[r][64];
      float se = 0.f, sw2 = 0.f;
#pragma unroll
      for (int j = 0; j < 16; ++j) {
        const float e = __expf(s_log[r][c0h + j] + b1r[c0h + j] - m);
        se += e;
        sw2 += e * w2r[c0h + j];
      }
      s_red[r][c] = se;
      s_red[r][32 + c] = sw2;
      __syncthreads();
      if (c == 0) {
        float a = 0.f, bsum = 0.f;
        for (int k = 0; k < 32; ++k) { a += s_red[r][k]; bsum += s_red[r][32 + k]; }
        const float g = bsum / a + op_b2[o];
        s_g[r][o] = 1.f / (1.f + __expf(-g));
      }
      __syncthreads();
    }

    // ---- H: new_control = tanh(relu(h0) @ W_control^T) ----
#pragma unroll
    for (int i = 0; i < 16; ++i) {
      const float v = s_h[r][c0h + i];
      s_abf[r][c0h + i] = f2bf(v > 0.f ? v : 0.f);
    }
    __syncthreads();
    gemm16(&s_abf[0][0], wct_bf, H_, &s_newc[0][0], V_, V_, H_, 0);
    __syncthreads();
#pragma unroll
    for (int i = 0; i < 8; ++i) s_newc[r][c0v + i] = tanhf(s_newc[r][c0v + i]);
    __syncthreads();

    // ---- I: work path: h_w = circ(circ(work_n, x_n), typ_n) @ sym_wn^T ----
    circconv256(&s_work[0][0], &s_x[0][0], &s_conv[0][0], V_, &s_dup[0][0]);
    __syncthreads();
    circconv256(&s_conv[0][0], &s_typ[0][0], &s_neww[0][0], V_, &s_dup[0][0]);
    __syncthreads();
#pragma unroll
    for (int i = 0; i < 8; ++i) s_abf[r][c0v + i] = f2bf(s_neww[r][c0v + i]);
    __syncthreads();
    gemm16(&s_abf[0][0], sym_bf, V_, &s_h[0][0], H_, H_, V_, 0);
    __syncthreads();
#pragma unroll
    for (int i = 0; i < 16; ++i) {
      const float v = s_h[r][c0h + i];
      s_abf[r][c0h + i] = f2bf(v > 0.f ? v : 0.f);
    }
    __syncthreads();
    gemm16(&s_abf[0][0], wwk_bf, H_, &s_neww[0][0], V_, V_, H_, 0);
    __syncthreads();
#pragma unroll
    for (int i = 0; i < 8; ++i) s_neww[r][c0v + i] = tanhf(s_neww[r][c0v + i]);
    __syncthreads();

    // ---- J: selector: h_s = relu(circ(circ(ctrl_n, work_n), typ_n) @ W_sel_n^T) ----
    circconv256(&s_ctrl[0][0], &s_work[0][0], &s_conv[0][0], V_, &s_dup[0][0]);
    __syncthreads();
    circconv256(&s_conv[0][0], &s_typ[0][0], &s_log[0][0], H_, &s_dup[0][0]);
    __syncthreads();
#pragma unroll
    for (int i = 0; i < 8; ++i) s_abf[r][c0v + i] = f2bf(s_log[r][c0v + i]);
    __syncthreads();
    gemm16(&s_abf[0][0], sel_bf, V_, &s_h[0][0], H_, H_, V_, 1);
    __syncthreads();
    if (tid < MB_ * 5) {
      const int rr = tid / 5, o = tid % 5;
      const float* wsr = W_sel2 + o * H_;
      float s = 0.f;
      for (int h = 0; h < H_; ++h) s += s_h[rr][h] * wsr[h];
      s_sm[rr][o] = s;
    }
    __syncthreads();
    if (tid < MB_) {
      float m = -1e30f;
#pragma unroll
      for (int o = 0; o < 5; ++o) m = fmaxf(m, s_sm[tid][o]);
      float e[5], se = 0.f;
#pragma unroll
      for (int o = 0; o < 5; ++o) { e[o] = __expf(s_sm[tid][o] - m); se += e[o]; }
      const float inv = 1.f / se;
#pragma unroll
      for (int o = 0; o < 5; ++o) s_sm[tid][o] = e[o] * inv;
    }
    __syncthreads();

    // ---- K: output (uses RAW pre-update work read) ----
    {
      const int bb = b0 + r;
      const float w0 = pw[bb * 4 + 0], w1 = pw[bb * 4 + 1], w2 = pw[bb * 4 + 2], w3 = pw[bb * 4 + 3];
#pragma unroll
      for (int i = 0; i < 8; ++i) {
        const int v = c0v + i;
        const float wr_raw = w0 * sw_[(size_t)(bb * 4 + 0) * V_ + v] + w1 * sw_[(size_t)(bb * 4 + 1) * V_ + v] +
                             w2 * sw_[(size_t)(bb * 4 + 2) * V_ + v] + w3 * sw_[(size_t)(bb * 4 + 3) * V_ + v];
        float val = s_sm[r][0] * wr_raw;
#pragma unroll
        for (int o = 0; o < 4; ++o) val += s_sm[r][1 + o] * out_sym[o * V_ + v];
        out[((size_t)bb * T_ + t) * V_ + v] = val;
      }
    }
    __syncthreads();

    // ---- L: soft stack updates ----
    if (tid < MB_) {
      const int bb = b0 + tid;
      const float shc = csharp[0], shw = wsharp[0];
      float p[4];
#pragma unroll
      for (int n = 0; n < 4; ++n) p[n] = pc[bb * 4 + n];
      {
        const float l0 = shc * s_g[tid][0], l1 = shc * s_g[tid][1], l2 = shc * s_g[tid][2];
        const float m = fmaxf(fmaxf(l0, l1), l2);
        const float e0 = __expf(l0 - m), e1 = __expf(l1 - m), e2 = __expf(l2 - m);
        const float inv = 1.f / (e0 + e1 + e2);
        const float ps = e0 * inv, pp = e1 * inv, pn = e2 * inv;
#pragma unroll
        for (int n = 0; n < 4; ++n) {
          const float qp = p[(n + 3) & 3];
          s_pt[tid][n] = p[n];
          s_pt[tid][4 + n] = qp;
          s_pt[tid][12 + n] = ps * qp + pp * p[(n + 1) & 3] + pn * p[n];
        }
        s_pt[tid][8] = ps; s_pt[tid][9] = pp; s_pt[tid][10] = pn;
      }
#pragma unroll
      for (int n = 0; n < 4; ++n) p[n] = pw[bb * 4 + n];
      {
        const float l0 = shw * s_g[tid][3], l1 = shw * s_g[tid][4], l2 = shw * s_g[tid][5];
        const float m = fmaxf(fmaxf(l0, l1), l2);
        const float e0 = __expf(l0 - m), e1 = __expf(l1 - m), e2 = __expf(l2 - m);
        const float inv = 1.f / (e0 + e1 + e2);
        const float ps = e0 * inv, pp = e1 * inv, pn = e2 * inv;
#pragma unroll
        for (int n = 0; n < 4; ++n) {
          const float qp = p[(n + 3) & 3];
          s_pt[tid][16 + n] = p[n];
          s_pt[tid][20 + n] = qp;
          s_pt[tid][28 + n] = ps * qp + pp * p[(n + 1) & 3] + pn * p[n];
        }
        s_pt[tid][24] = ps; s_pt[tid][25] = pp; s_pt[tid][26] = pn;
      }
    }
    __syncthreads();
    for (int idx = tid; idx < MB_ * NS_ * V_; idx += NT_) {
      const int rr = idx >> 10;
      const int n  = (idx >> 8) & 3;
      const int v  = idx & 255;
      const size_t off = (size_t)((b0 + rr) * 4 + n) * V_ + v;
      {
        const float old = sc[off];
        const float qp = s_pt[rr][4 + n], pt = s_pt[rr][n];
        const float ps = s_pt[rr][8], pp = s_pt[rr][9], pn = s_pt[rr][10];
        const float pushv = old * (1.f - qp) + s_newc[rr][v] * qp;
        const float popv  = old * (1.f - pt) + ZO_ * pt;
        sc[off] = ps * pushv + pp * popv + pn * old;
      }
      {
        const float old = sw_[off];
        const float qp = s_pt[rr][20 + n], pt = s_pt[rr][16 + n];
        const float ps = s_pt[rr][24], pp = s_pt[rr][25], pn = s_pt[rr][26];
        const float pushv = old * (1.f - qp) + s_neww[rr][v] * qp;
        const float popv  = old * (1.f - pt) + ZO_ * pt;
        sw_[off] = ps * pushv + pp * popv + pn * old;
      }
    }
    __syncthreads();
    if (tid < MB_) {
      const int bb = b0 + tid;
#pragma unroll
      for (int n = 0; n < 4; ++n) {
        pc[bb * 4 + n] = s_pt[tid][12 + n];
        pw[bb * 4 + n] = s_pt[tid][28 + n];
      }
    }
    __syncthreads();
  }
}

// ---------------- launch ----------------
extern "C" void kernel_launch(void* const* d_in, const int* in_sizes, int n_in,
                              void* d_out, int out_size, void* d_ws, size_t ws_size,
                              hipStream_t stream) {
  (void)in_sizes; (void)n_in; (void)out_size; (void)ws_size;
  const float* x        = (const float*)d_in[0];
  const float* sym_w    = (const float*)d_in[1];
  const float* W_isa    = (const float*)d_in[2];
  const float* W_isa2   = (const float*)d_in[3];
  const float* op_W1    = (const float*)d_in[4];
  const float* op_b1    = (const float*)d_in[5];
  const float* op_W2    = (const float*)d_in[6];
  const float* op_b2    = (const float*)d_in[7];
  const float* W_ctrl   = (const float*)d_in[8];
  const float* W_work   = (const float*)d_in[9];
  const float* W_seln   = (const float*)d_in[10];
  const float* W_sel2   = (const float*)d_in[11];
  const float* out_sym  = (const float*)d_in[12];
  const float* initvec  = (const float*)d_in[13];
  const float* csharp   = (const float*)d_in[14];
  const float* wsharp   = (const float*)d_in[15];
  float* out = (float*)d_out;

  char* ws = (char*)d_ws;
  const size_t o_sym = 0;
  const size_t o_sel = o_sym + (size_t)H_ * V_ * 2;
  const size_t o_opw = o_sel + (size_t)H_ * V_ * 2;
  const size_t o_wct = o_opw + (size_t)6 * H_ * H_ * 2;
  const size_t o_wwk = o_wct + (size_t)V_ * H_ * 2;
  const size_t o_isa = o_wwk + (size_t)V_ * H_ * 2;
  const size_t o_sc  = o_isa + (size_t)4 * V_ * 4;
  const size_t o_sw  = o_sc + (size_t)B_ * NS_ * V_ * 4;
  const size_t o_pc  = o_sw + (size_t)B_ * NS_ * V_ * 4;
  const size_t o_pw  = o_pc + (size_t)B_ * NS_ * 4;

  uint16_t* sym_bf  = (uint16_t*)(ws + o_sym);
  uint16_t* sel_bf  = (uint16_t*)(ws + o_sel);
  uint16_t* opw1_bf = (uint16_t*)(ws + o_opw);
  uint16_t* wct_bf  = (uint16_t*)(ws + o_wct);
  uint16_t* wwk_bf  = (uint16_t*)(ws + o_wwk);
  float* wisa_n = (float*)(ws + o_isa);
  float* sc = (float*)(ws + o_sc);
  float* sw = (float*)(ws + o_sw);
  float* pc = (float*)(ws + o_pc);
  float* pw = (float*)(ws + o_pw);

  rownorm_bf16_k<<<H_, 256, 0, stream>>>(sym_w, sym_bf);
  rownorm_bf16_k<<<H_, 256, 0, stream>>>(W_seln, sel_bf);
  rownorm_f32_k<<<4, 256, 0, stream>>>(W_isa, wisa_n);
  cvt_bf16_k<<<2048, 256, 0, stream>>>(op_W1, opw1_bf, 6 * H_ * H_);
  cvt_bf16_k<<<512, 256, 0, stream>>>(W_ctrl, wct_bf, V_ * H_);
  cvt_bf16_k<<<512, 256, 0, stream>>>(W_work, wwk_bf, V_ * H_);
  stack_init_k<<<(B_ * NS_ * V_ + 255) / 256, 256, 0, stream>>>(initvec, sc, sw, pc, pw);

  ns_main<<<B_ / MB_, NT_, 0, stream>>>(
      x, W_isa2, op_b1, op_W2, op_b2, W_sel2, out_sym, csharp, wsharp,
      sym_bf, sel_bf, opw1_bf, wct_bf, wwk_bf, wisa_n,
      sc, sw, pc, pw, out);
}
